// EBEPCETModelWrapper_29119878267041
// MI455X (gfx1250) — matrix operand order repaired and checked by reading, not compiled, not run on hardware
//
#include <hip/hip_runtime.h>
#include <hip/hip_bf16.h>

// MI455X (gfx1250) energy-model kernel, round 2.
// All contractions on v_wmma_f32_16x16x32_bf16 (wave32, fp32 accum).
// Round-2 changes: 16B-aligned LDS strides -> ds_load_b128 fragments;
// dual independent accumulation chains per wave (fills WMMA hazard slots);
// software-pipelined B fragments; z/patch A-fragments hoisted to VGPRs.

typedef __attribute__((ext_vector_type(16))) __bf16 v16bf;
typedef __attribute__((ext_vector_type(8)))  float  v8f;

#define BDIM 256

constexpr int DP    = 256;   // D
constexpr int NPAT  = 64;    // NP
constexpr int MMEM  = 1024;  // M
constexpr int KENC  = 192;   // C*P*P
constexpr float GAMMA_C = 0.25f;

// LDS partition (32-bit words), pair-packed bf16. Row strides are multiples of
// 4 words (16B) so fragments load as ds_load_b128; stride%64 != 0 keeps the
// 16-lane row gathers on distinct banks (132%64=4, 100%64=36).
constexpr int SZ    = 132;                   // 264 bf16 per row
constexpr int SP    = 100;                   // 200 bf16 per row (patches)
constexpr int Z_OFF = 0;                     // z bf16        64 x SZ
constexpr int P_OFF = Z_OFF + 64 * SZ;       // patches bf16  64 x SP
constexpr int W_OFF = P_OFF + 64 * SP;       // B-panel bf16  64 x SZ (reused)
constexpr int Q_OFF = W_OFF + 64 * SZ;       // Q bf16        64 x SZ
constexpr int K_OFF = Q_OFF + 64 * SZ;       // K bf16        64 x SZ
constexpr int R_OFF = K_OFF + 64 * SZ;       // reduction     256 floats
constexpr int SMEM_WORDS = R_OFF + 256;
constexpr size_t SMEM_BYTES = (size_t)SMEM_WORDS * 4;  // 161792 B (~158 KB)

union FragU { v16bf v; unsigned u[8]; uint4 q[2]; };

__device__ __forceinline__ unsigned pack_bf2(float a, float b) {
  union { float f; unsigned u; } ua, ub; ua.f = a; ub.f = b;
  unsigned x = ua.u, y = ub.u;
  x = (x + 0x7FFFu + ((x >> 16) & 1u)) >> 16;   // RNE
  y = (y + 0x7FFFu + ((y >> 16) & 1u)) >> 16;
  return (y << 16) | (x & 0xFFFFu);
}
__device__ __forceinline__ unsigned short f2bf(float f) {
  union { float f; unsigned u; } v; v.f = f;
  return (unsigned short)((v.u + 0x7FFFu + ((v.u >> 16) & 1u)) >> 16);
}
__device__ __forceinline__ float bf2f(unsigned short h) {
  union { unsigned u; float f; } v; v.u = ((unsigned)h) << 16; return v.f;
}

// A fragment (16x32 bf16). ISA layout: lanes0-15 V0..3->K0..7, V4..7->K16..23;
// lanes16-31 +8. Two contiguous 16B chunks per lane.
__device__ __forceinline__ v16bf load_frag_A(const unsigned* base, int strideW,
                                             int mRow, int kBase) {
  const int lane = threadIdx.x & 31;
  const unsigned* p = base + (mRow + (lane & 15)) * strideW
                    + (kBase >> 1) + ((lane >> 4) << 2);
  FragU f;
  f.q[0] = *(const uint4*)p;        // K pairs kBase+hi*8+{0,2,4,6}
  f.q[1] = *(const uint4*)(p + 8);  // K pairs +16
  return f.v;
}

// B fragment (32x16 bf16) from [n][k] pair-packed storage:
// lanes0-15 V0..3->K0..7, V4..7->K16..23; lanes16-31 +8. Two 16B chunks per lane.
__device__ __forceinline__ v16bf load_frag_B(const unsigned* base, int strideW,
                                             int nCol, int kBase) {
  const int lane = threadIdx.x & 31;
  const unsigned* p = base + (nCol + (lane & 15)) * strideW
                    + (kBase >> 1) + ((lane >> 4) << 2);
  FragU f;
  f.q[0] = *(const uint4*)p;
  f.q[1] = *(const uint4*)(p + 8);
  return f.v;
}

__device__ __forceinline__ v8f wmma_bf16(v16bf a, v16bf b, v8f c) {
  return __builtin_amdgcn_wmma_f32_16x16x32_bf16(false, a, false, b,
                                                 (short)0, c, false, false);
}

__global__ __launch_bounds__(BDIM, 1)
void energy_kernel(const float* __restrict__ x,     const float* __restrict__ z,
                   const float* __restrict__ enc_w, const float* __restrict__ enc_b,
                   const float* __restrict__ vis_b, const float* __restrict__ pos_b,
                   const float* __restrict__ mem_w, const float* __restrict__ wq,
                   const float* __restrict__ wk,    float* __restrict__ out) {
  extern __shared__ unsigned smem[];
  unsigned short* smem16 = (unsigned short*)smem;

  const int t = threadIdx.x;
  const int lane = t & 31, wave = t >> 5;
  const int hi = lane >> 4, l16 = lane & 15;
  const int b = blockIdx.x;

  // Per-wave tile assignment: tiles (mt, nt0) and (mt, nt1) share the A row
  // block -> one A fragment feeds two independent accumulation chains.
  const int mt  = wave & 3;
  const int nt0 = wave >> 2;        // 0 or 1
  const int nt1 = nt0 + 2;          // 2 or 3

  float acc = 0.f;  // E accumulator: quad terms +, coupling terms -

  // ---------------- Phase 0: z -> LDS bf16 + z scalar terms ----------------
  const float* zb = z + (size_t)b * NPAT * DP;
#pragma unroll 4
  for (int i = 0; i < 32; ++i) {
    int pi = t + i * BDIM;            // 8192 bf16-pairs
    int p = pi >> 7, dp = pi & 127;
    float z0 = zb[p * DP + 2 * dp];
    float z1 = zb[p * DP + 2 * dp + 1];
    smem[Z_OFF + p * SZ + dp] = pack_bf2(z0, z1);
    int d0 = 2 * dp;
    acc += 0.5f * (z0 * z0 + z1 * z1)
         - z0 * (enc_b[d0]     + pos_b[p * DP + d0])
         - z1 * (enc_b[d0 + 1] + pos_b[p * DP + d0 + 1]);
  }
  // x -> patch matrix (64x192 bf16) + quad_x + phi_vis
  const float* xb = x + (size_t)b * 3 * 64 * 64;
#pragma unroll 4
  for (int i = 0; i < 48; ++i) {
    int idx = t + i * BDIM;           // 12288 elements
    int p = idx / KENC, k = idx % KENC;
    int c = k >> 6, r = (k & 63) >> 3, col = k & 7;
    int gy = p >> 3, gx = p & 7;
    int off = c * 4096 + (gy * 8 + r) * 64 + gx * 8 + col;
    float xv = xb[off];
    acc += 0.5f * xv * xv - xv * vis_b[off];
    smem16[(P_OFF << 1) + p * (SP * 2) + k] = f2bf(xv);
  }
  __syncthreads();

  // ---------------- Phase 1: conv-encode GEMM + fused z-dot ----------------
  {
    FragU ap[6];                      // patch A-fragments, K=192 -> 6 steps
#pragma unroll
    for (int kb = 0; kb < 6; ++kb)
      ap[kb].v = load_frag_A(smem + P_OFF, SP, mt * 16, kb * 32);

    for (int pn = 0; pn < 4; ++pn) {
      const int P0 = pn * 64;
#pragma unroll 4
      for (int i = 0; i < 24; ++i) {
        int pi = t + i * BDIM;        // 6144 pairs
        int n = pi / 96, kp = pi % 96;
        const float* src = enc_w + (size_t)(P0 + n) * KENC + 2 * kp;
        smem[W_OFF + n * SZ + kp] = pack_bf2(src[0], src[1]);
      }
      __syncthreads();
      v8f c0 = {}, c1 = {};
      FragU b0, b1, n0, n1;
      b0.v = load_frag_B(smem + W_OFF, SZ, nt0 * 16, 0);
      b1.v = load_frag_B(smem + W_OFF, SZ, nt1 * 16, 0);
#pragma unroll
      for (int kb = 0; kb < 6; ++kb) {
        if (kb < 5) {
          n0.v = load_frag_B(smem + W_OFF, SZ, nt0 * 16, (kb + 1) * 32);
          n1.v = load_frag_B(smem + W_OFF, SZ, nt1 * 16, (kb + 1) * 32);
        }
        c0 = wmma_bf16(ap[kb].v, b0.v, c0);
        c1 = wmma_bf16(ap[kb].v, b1.v, c1);
        b0 = n0; b1 = n1;
      }
#pragma unroll
      for (int r = 0; r < 8; ++r) {
        int m = mt * 16 + r + hi * 8;
        int na = P0 + nt0 * 16 + l16;
        int nb = P0 + nt1 * 16 + l16;
        acc -= c0[r] * bf2f(smem16[(Z_OFF << 1) + m * (SZ * 2) + na]);
        acc -= c1[r] * bf2f(smem16[(Z_OFF << 1) + m * (SZ * 2) + nb]);
      }
      __syncthreads();
    }
  }

  // z A-fragments: invariant for phases 2 and 4 -> hoist into 64 VGPRs.
  FragU az[8];
#pragma unroll
  for (int kb = 0; kb < 8; ++kb)
    az[kb].v = load_frag_A(smem + Z_OFF, SZ, mt * 16, kb * 32);

  // ---------------- Phase 2: Q and K projections ---------------------------
  for (int which = 0; which < 2; ++which) {
    const float* W = which ? wk : wq;
    const int dstOff = which ? K_OFF : Q_OFF;
    for (int pn = 0; pn < 4; ++pn) {
      const int P0 = pn * 64;
#pragma unroll 4
      for (int i = 0; i < 32; ++i) {
        int pi = t + i * BDIM;        // 8192 pairs
        int n = pi >> 7, kp = pi & 127;
        const float* src = W + (size_t)(P0 + n) * DP + 2 * kp;
        smem[W_OFF + n * SZ + kp] = pack_bf2(src[0], src[1]);
      }
      __syncthreads();
      v8f c0 = {}, c1 = {};
      FragU b0, b1, n0, n1;
      b0.v = load_frag_B(smem + W_OFF, SZ, nt0 * 16, 0);
      b1.v = load_frag_B(smem + W_OFF, SZ, nt1 * 16, 0);
#pragma unroll
      for (int kb = 0; kb < 8; ++kb) {
        if (kb < 7) {
          n0.v = load_frag_B(smem + W_OFF, SZ, nt0 * 16, (kb + 1) * 32);
          n1.v = load_frag_B(smem + W_OFF, SZ, nt1 * 16, (kb + 1) * 32);
        }
        c0 = wmma_bf16(az[kb].v, b0.v, c0);
        c1 = wmma_bf16(az[kb].v, b1.v, c1);
        b0 = n0; b1 = n1;
      }
#pragma unroll
      for (int r = 0; r < 8; ++r) {
        int m = mt * 16 + r + hi * 8;
        smem16[(dstOff << 1) + m * (SZ * 2) + P0 + nt0 * 16 + l16] = f2bf(c0[r]);
        smem16[(dstOff << 1) + m * (SZ * 2) + P0 + nt1 * 16 + l16] = f2bf(c1[r]);
      }
      __syncthreads();
    }
  }

  // ---------------- Phase 3: attention scores + logsumexp ------------------
  // One wave per head; 4 independent chains (nt=0..3) share each A fragment.
  {
    const int h = wave;               // NH == 8 == #waves
    FragU kb0, kb1, kb2, kb3;         // per-head B fragments, mt-invariant
    kb0.v = load_frag_B(smem + K_OFF, SZ,  0, h * 32);
    kb1.v = load_frag_B(smem + K_OFF, SZ, 16, h * 32);
    kb2.v = load_frag_B(smem + K_OFF, SZ, 32, h * 32);
    kb3.v = load_frag_B(smem + K_OFF, SZ, 48, h * 32);
    for (int amt = 0; amt < 4; ++amt) {
      v16bf a = load_frag_A(smem + Q_OFF, SZ, amt * 16, h * 32);
      v8f c0 = {}, c1 = {}, c2 = {}, c3 = {};
      c0 = wmma_bf16(a, kb0.v, c0);
      c1 = wmma_bf16(a, kb1.v, c1);
      c2 = wmma_bf16(a, kb2.v, c2);
      c3 = wmma_bf16(a, kb3.v, c3);
#pragma unroll
      for (int r = 0; r < 8; ++r) {
        float t0 = GAMMA_C * c0[r], t1 = GAMMA_C * c1[r];
        float t2 = GAMMA_C * c2[r], t3 = GAMMA_C * c3[r];
        float mx = fmaxf(fmaxf(t0, t1), fmaxf(t2, t3));
#pragma unroll
        for (int sm = 1; sm < 16; sm <<= 1) mx = fmaxf(mx, __shfl_xor(mx, sm, 16));
        float s = __expf(t0 - mx) + __expf(t1 - mx) + __expf(t2 - mx) + __expf(t3 - mx);
#pragma unroll
        for (int sm = 1; sm < 16; sm <<= 1) s += __shfl_xor(s, sm, 16);
        if (l16 == 0) acc -= (mx + __logf(s)) * 4.0f;   // 1/GAMMA
      }
    }
  }
  __syncthreads();

  // ---------------- Phase 4: memory GEMM + fused relu^2 --------------------
  for (int pn = 0; pn < 16; ++pn) {
    const int N0 = pn * 64;
#pragma unroll 4
    for (int i = 0; i < 32; ++i) {
      int pi = t + i * BDIM;          // 8192 pairs; pairs along k (row) dim
      int n = pi & 63, kp = pi >> 6;
      float w0 = mem_w[(size_t)(2 * kp)     * MMEM + N0 + n];
      float w1 = mem_w[(size_t)(2 * kp + 1) * MMEM + N0 + n];
      smem[W_OFF + n * SZ + kp] = pack_bf2(w0, w1);
    }
    if (pn + 1 < 16)  // global_prefetch of next panel (one line per row)
      __builtin_prefetch(mem_w + (size_t)t * MMEM + N0 + 64, 0, 1);
    __syncthreads();
    v8f c0 = {}, c1 = {};
    FragU b0, b1, n0, n1;
    b0.v = load_frag_B(smem + W_OFF, SZ, nt0 * 16, 0);
    b1.v = load_frag_B(smem + W_OFF, SZ, nt1 * 16, 0);
#pragma unroll
    for (int kb = 0; kb < 8; ++kb) {
      if (kb < 7) {
        n0.v = load_frag_B(smem + W_OFF, SZ, nt0 * 16, (kb + 1) * 32);
        n1.v = load_frag_B(smem + W_OFF, SZ, nt1 * 16, (kb + 1) * 32);
      }
      c0 = wmma_bf16(az[kb].v, b0.v, c0);
      c1 = wmma_bf16(az[kb].v, b1.v, c1);
      b0 = n0; b1 = n1;
    }
#pragma unroll
    for (int r = 0; r < 8; ++r) {
      float v0 = c0[r] > 0.f ? c0[r] : 0.f;
      float v1 = c1[r] > 0.f ? c1[r] : 0.f;
      acc -= v0 * v0 + v1 * v1;
    }
    __syncthreads();
  }

  // ---------------- Phase 5: block reduction -> out[b] ---------------------
  float* red = (float*)(smem + R_OFF);
  red[t] = acc;
  __syncthreads();
#pragma unroll
  for (int s = BDIM / 2; s > 0; s >>= 1) {
    if (t < s) red[t] += red[t + s];
    __syncthreads();
  }
  if (t == 0) out[b] = red[0];
}

extern "C" void kernel_launch(void* const* d_in, const int* in_sizes, int n_in,
                              void* d_out, int out_size, void* d_ws, size_t ws_size,
                              hipStream_t stream) {
  const float* x     = (const float*)d_in[0];
  const float* z     = (const float*)d_in[1];
  const float* enc_w = (const float*)d_in[2];
  const float* enc_b = (const float*)d_in[3];
  const float* vis_b = (const float*)d_in[4];
  const float* pos_b = (const float*)d_in[5];
  const float* mem_w = (const float*)d_in[6];
  const float* wq    = (const float*)d_in[7];
  const float* wk    = (const float*)d_in[8];
  float* out = (float*)d_out;

  const int B = in_sizes[0] / (3 * 64 * 64);

  static bool attr_set = false;  // host-side only; idempotent
  if (!attr_set) {
    (void)hipFuncSetAttribute((const void*)energy_kernel,
                              hipFuncAttributeMaxDynamicSharedMemorySize,
                              (int)SMEM_BYTES);
    attr_set = true;
  }

  energy_kernel<<<dim3(B), dim3(BDIM), SMEM_BYTES, stream>>>(
      x, z, enc_w, enc_b, vis_b, pos_b, mem_w, wq, wk, out);
}
